// QLSTMPlus_65481071408563
// MI455X (gfx1250) — matrix operand order repaired and checked by reading, not compiled, not run on hardware
//
#include <hip/hip_runtime.h>
#include <hip/hip_bf16.h>
#include <math.h>

// ---------------------------------------------------------------------------
// QLSTM on MI455X (gfx1250): wave32 WMMA f16->f32 GEMMs, 2 kernels per step.
// T=256, B=128, D=512, H=512.  Gates packed as N = 4*H = 2048 columns.
// x is pre-converted to f16 once; recurrent GEMMs are pure f16 + WMMA.
// ---------------------------------------------------------------------------

#define TT 256
#define BB 128
#define DD 512
#define HH 512
#define NG 2048   // 4*HH

typedef __attribute__((ext_vector_type(16))) _Float16 v16h;
typedef __attribute__((ext_vector_type(8)))  float    v8f;
typedef __attribute__((ext_vector_type(4)))  float    f32x4;

// ---- WMMA fragment loaders (CDNA5 16x16x32 f16 layouts) -------------------

// A-matrix 16x32: lane l -> row m=l&15; K = [8h,8h+8) U [16+8h,16+8h+8)
__device__ __forceinline__ v16h fragA_f16(const _Float16* __restrict__ row, int kBase) {
  const int half = (threadIdx.x >> 4) & 1;
  const _Float16* p = row + kBase + half * 8;
  union { v16h v; f32x4 q[2]; } u;
  u.q[0] = *(const f32x4*)(p);        // K: kBase+8h .. +7
  u.q[1] = *(const f32x4*)(p + 16);   // K: kBase+16+8h .. +7
  return u.v;
}

// B-matrix 32x16: lane l -> col n=l&15; K = [8h,8h+8) U [16+8h,16+8h+8).
// `row` points at row n of B^T (i.e. column n of B), so two 16B loads.
__device__ __forceinline__ v16h fragB_f16(const _Float16* __restrict__ row, int kBase) {
  const int half = (threadIdx.x >> 4) & 1;
  const _Float16* p = row + kBase + half * 8;
  union { v16h v; f32x4 q[2]; } u;
  u.q[0] = *(const f32x4*)(p);
  u.q[1] = *(const f32x4*)(p + 16);
  return u.v;
}

// ---------------------------------------------------------------------------
// One-time prep kernels
// ---------------------------------------------------------------------------

// Whole input tensor fp32 -> f16 (T*B*D elements), once per call.
__global__ void qlstm_cvt_x(const float* __restrict__ x, _Float16* __restrict__ xh,
                            int n) {
  int idx = blockIdx.x * 256 + threadIdx.x;
  if (idx < n) xh[idx] = (_Float16)x[idx];
}

// Wt[g][n][k] = (f16) W_g[k][n]   (W_g is (D+H, H) row-major)
__global__ void qlstm_pack_w(const float* __restrict__ W0, const float* __restrict__ W1,
                             const float* __restrict__ W2, const float* __restrict__ W3,
                             _Float16* __restrict__ Wt) {
  int idx = blockIdx.x * 256 + threadIdx.x;          // 4*512*1024
  if (idx >= 4 * HH * (DD + HH)) return;
  int g   = idx >> 19;                               // 512*1024 = 2^19
  int rem = idx & ((1 << 19) - 1);
  int n   = rem >> 10;
  int k   = rem & 1023;
  const float* W = (g == 0) ? W0 : (g == 1) ? W1 : (g == 2) ? W2 : W3;
  Wt[idx] = (_Float16)W[(size_t)k * HH + n];
}

// Pt[g][n][k] = (f16) P_g[k][n]   (P_g is (H, H) row-major)
__global__ void qlstm_pack_p(const float* __restrict__ P0, const float* __restrict__ P1,
                             const float* __restrict__ P2, const float* __restrict__ P3,
                             _Float16* __restrict__ Pt) {
  int idx = blockIdx.x * 256 + threadIdx.x;          // 4*512*512
  if (idx >= 4 * HH * HH) return;
  int g   = idx >> 18;                               // 512*512 = 2^18
  int rem = idx & ((1 << 18) - 1);
  int n   = rem >> 9;
  int k   = rem & 511;
  const float* P = (g == 0) ? P0 : (g == 1) ? P1 : (g == 2) ? P2 : P3;
  Pt[idx] = (_Float16)P[(size_t)k * HH + n];
}

__global__ void qlstm_pack_misc(const float* __restrict__ b0, const float* __restrict__ b1,
                                const float* __restrict__ b2, const float* __restrict__ b3,
                                const float* __restrict__ r0, const float* __restrict__ r1,
                                const float* __restrict__ r2, const float* __restrict__ r3,
                                float* __restrict__ biasAll, float* __restrict__ cosAll) {
  int idx = blockIdx.x * 256 + threadIdx.x;          // 2048
  if (idx >= NG) return;
  int g = idx >> 9, j = idx & 511;
  const float* b = (g == 0) ? b0 : (g == 1) ? b1 : (g == 2) ? b2 : b3;
  const float* r = (g == 0) ? r0 : (g == 1) ? r1 : (g == 2) ? r2 : r3;
  biasAll[idx] = b[j];
  cosAll[idx]  = cosf(r[j]);
}

__global__ void qlstm_init_state(_Float16* __restrict__ hbuf, float* __restrict__ cbuf) {
  int idx = blockIdx.x * 256 + threadIdx.x;          // 128*512
  if (idx >= BB * HH) return;
  hbuf[idx] = (_Float16)0.0f;
  cbuf[idx] = 0.0f;
}

// ---------------------------------------------------------------------------
// Kernel A: z' = ((comb @ W) + b) * cos(rx), comb = [x_t (f16), h (f16)]
// M=128, N=2048, K=1024.  Each wave computes a 16x32 strip (two N tiles),
// reusing the A fragment across both B fragments. 512 waves = 64 WGs.
// ---------------------------------------------------------------------------
__global__ void __launch_bounds__(256)
qlstm_gates_z(const _Float16* __restrict__ xt,      // (B, D) f16, = xh + t*B*D
              const _Float16* __restrict__ hbuf,    // (B, H) f16
              const _Float16* __restrict__ Wt,      // (4,512,1024) f16 (W^T rows)
              const float* __restrict__ biasAll,    // (2048)
              const float* __restrict__ cosAll,     // (2048)
              _Float16* __restrict__ zbuf)          // (B, 2048) f16
{
  const int lane = threadIdx.x & 31;
  const int id   = blockIdx.x * 8 + (threadIdx.x >> 5);   // 0..511
  const int mt   = id >> 6;                               // 0..7
  const int ntp  = id & 63;                               // 0..63 (pair of N tiles)
  const int m    = mt * 16 + (lane & 15);
  const int col0 = ntp * 32 + (lane & 15);                // first tile column
  const int col1 = col0 + 16;                             // second tile column

  const _Float16* xrow  = xt   + (size_t)m * DD;
  const _Float16* hrow  = hbuf + (size_t)m * HH;
  const _Float16* wrow0 = Wt + (size_t)col0 * (DD + HH);
  const _Float16* wrow1 = Wt + (size_t)col1 * (DD + HH);

  v8f acc0 = {}, acc1 = {};
#pragma unroll 4
  for (int k = 0; k < DD; k += 32) {                      // x-part, K 0..511
    v16h a  = fragA_f16(xrow, k);
    v16h b0 = fragB_f16(wrow0, k);
    v16h b1 = fragB_f16(wrow1, k);
    acc0 = __builtin_amdgcn_wmma_f32_16x16x32_f16(false, a, false, b0,
                                                  (short)0, acc0, false, false);
    acc1 = __builtin_amdgcn_wmma_f32_16x16x32_f16(false, a, false, b1,
                                                  (short)0, acc1, false, false);
  }
#pragma unroll 4
  for (int k = 0; k < HH; k += 32) {                      // h-part, K 512..1023
    v16h a  = fragA_f16(hrow, k);
    v16h b0 = fragB_f16(wrow0, DD + k);
    v16h b1 = fragB_f16(wrow1, DD + k);
    acc0 = __builtin_amdgcn_wmma_f32_16x16x32_f16(false, a, false, b0,
                                                  (short)0, acc0, false, false);
    acc1 = __builtin_amdgcn_wmma_f32_16x16x32_f16(false, a, false, b1,
                                                  (short)0, acc1, false, false);
  }

  const int   half   = lane >> 4;
  const float bias0  = biasAll[col0], scale0 = cosAll[col0];
  const float bias1  = biasAll[col1], scale1 = cosAll[col1];
#pragma unroll
  for (int r = 0; r < 8; ++r) {
    int row = mt * 16 + r + 8 * half;
    zbuf[(size_t)row * NG + col0] = (_Float16)((acc0[r] + bias0) * scale0);
    zbuf[(size_t)row * NG + col1] = (_Float16)((acc1[r] + bias1) * scale1);
  }
}

// ---------------------------------------------------------------------------
// Kernel B: a = z'_g @ P_g, activations, LSTM combine (all 4 gates in-WG).
// Each WG: 8 waves = 2 tile positions x 4 gates; LDS combine.
// ---------------------------------------------------------------------------
__global__ void __launch_bounds__(256)
qlstm_gates_combine(const _Float16* __restrict__ zbuf,  // (B, 2048) f16
                    const _Float16* __restrict__ Pt,    // (4,512,512) f16 (P^T rows)
                    float* __restrict__ cbuf,           // (B, H) fp32
                    _Float16* __restrict__ hbuf,        // (B, H) f16
                    float* __restrict__ out,            // outputs|hx|cx fp32
                    int t, int isLast)
{
  __shared__ float lds[2][4][16][16];

  const int lane = threadIdx.x & 31;
  const int w    = threadIdx.x >> 5;
  const int p    = w >> 2;                      // 0..1 position within WG
  const int g    = w & 3;                       // gate
  const int pos  = blockIdx.x * 2 + p;          // 0..255
  const int mt   = pos & 7;                     // batch tile
  const int jt   = pos >> 3;                    // hidden tile 0..31
  const int m    = mt * 16 + (lane & 15);
  const int j    = jt * 16 + (lane & 15);

  const _Float16* arow = zbuf + (size_t)m * NG + g * HH;
  const _Float16* brow = Pt + ((size_t)g * HH + j) * HH;

  v8f acc = {};
#pragma unroll 4
  for (int k = 0; k < HH; k += 32) {
    v16h a = fragA_f16(arow, k);
    v16h b = fragB_f16(brow, k);
    acc = __builtin_amdgcn_wmma_f32_16x16x32_f16(false, a, false, b,
                                                 (short)0, acc, false, false);
  }

  const int half = lane >> 4;
#pragma unroll
  for (int r = 0; r < 8; ++r) {
    float v = acc[r];
    v = (g == 2) ? tanhf(v) : 1.0f / (1.0f + __expf(-v));   // u -> tanh, else sigmoid
    lds[p][g][r + 8 * half][lane & 15] = v;
  }
  __syncthreads();

  // Elementwise LSTM update: 2 positions x 256 elems = 512, 2 per thread.
  for (int e = threadIdx.x; e < 512; e += 256) {
    int pp   = e >> 8;
    int idx  = e & 255;
    int ml   = idx >> 4, nl = idx & 15;
    int posE = blockIdx.x * 2 + pp;
    int b    = (posE & 7) * 16 + ml;            // batch index
    int n    = (posE >> 3) * 16 + nl;           // hidden index
    float fg = lds[pp][0][ml][nl];
    float ig = lds[pp][1][ml][nl];
    float ug = lds[pp][2][ml][nl];
    float og = lds[pp][3][ml][nl];
    size_t ci = (size_t)b * HH + n;
    float c = fg * cbuf[ci] + ig * ug;
    cbuf[ci] = c;
    float h = og * tanhf(c);
    hbuf[ci] = (_Float16)h;
    out[(size_t)t * (BB * HH) + ci] = h;
    if (isLast) {
      out[(size_t)TT * BB * HH + ci] = h;                 // hx
      out[(size_t)TT * BB * HH + BB * HH + ci] = c;       // cx
    }
  }
}

// ---------------------------------------------------------------------------
extern "C" void kernel_launch(void* const* d_in, const int* in_sizes, int n_in,
                              void* d_out, int out_size, void* d_ws, size_t ws_size,
                              hipStream_t stream) {
  (void)in_sizes; (void)n_in; (void)out_size; (void)ws_size;

  const float* x = (const float*)d_in[0];
  // gate order f,i,u,o: W=1+4j, b=2+4j, rx=3+4j, P=4+4j
  const float* W[4]; const float* bG[4]; const float* rx[4]; const float* P[4];
  for (int jg = 0; jg < 4; ++jg) {
    W[jg]  = (const float*)d_in[1 + 4 * jg];
    bG[jg] = (const float*)d_in[2 + 4 * jg];
    rx[jg] = (const float*)d_in[3 + 4 * jg];
    P[jg]  = (const float*)d_in[4 + 4 * jg];
  }
  float* out = (float*)d_out;

  // Workspace layout (~40.8 MB)
  char* ws = (char*)d_ws;
  _Float16* xh      = (_Float16*)(ws);                 // T*B*D*2 = 33,554,432
  _Float16* Wt      = (_Float16*)(ws + 33554432);      // 4,194,304
  _Float16* Pt      = (_Float16*)(ws + 37748736);      // 2,097,152
  float*    biasAll = (float*)   (ws + 39845888);      // 8,192
  float*    cosAll  = (float*)   (ws + 39854080);      // 8,192
  _Float16* hbuf    = (_Float16*)(ws + 39862272);      // 131,072
  float*    cbuf    = (float*)   (ws + 39993344);      // 262,144
  _Float16* zbuf    = (_Float16*)(ws + 40255488);      // 524,288

  // One-time packing / init (deterministic each call)
  const int NX = TT * BB * DD;
  qlstm_cvt_x<<<(NX + 255) / 256, 256, 0, stream>>>(x, xh, NX);
  qlstm_pack_w<<<(4 * HH * (DD + HH) + 255) / 256, 256, 0, stream>>>(
      W[0], W[1], W[2], W[3], Wt);
  qlstm_pack_p<<<(4 * HH * HH + 255) / 256, 256, 0, stream>>>(
      P[0], P[1], P[2], P[3], Pt);
  qlstm_pack_misc<<<(NG + 255) / 256, 256, 0, stream>>>(
      bG[0], bG[1], bG[2], bG[3], rx[0], rx[1], rx[2], rx[3], biasAll, cosAll);
  qlstm_init_state<<<(BB * HH + 255) / 256, 256, 0, stream>>>(hbuf, cbuf);

  // Recurrent chain: 2 kernels per timestep
  for (int t = 0; t < TT; ++t) {
    const _Float16* xt = xh + (size_t)t * BB * DD;
    qlstm_gates_z<<<64, 256, 0, stream>>>(xt, hbuf, Wt, biasAll, cosAll, zbuf);
    qlstm_gates_combine<<<128, 256, 0, stream>>>(zbuf, Pt, cbuf, hbuf, out,
                                                 t, (t == TT - 1) ? 1 : 0);
  }
}
